// ModelMultitaskBinary_19370302505077
// MI455X (gfx1250) — compile-verified, matrix operand order checked by reading, not run on hardware
//
#include <hip/hip_runtime.h>
#include <hip/hip_bf16.h>
#include <math.h>

typedef __attribute__((ext_vector_type(16))) _Float16 v16h;
typedef __attribute__((ext_vector_type(8)))  _Float16 v8h;
typedef __attribute__((ext_vector_type(8)))  float    v8f;

// ---------------- problem dimensions ----------------
static constexpr int kB  = 512;
static constexpr int kC  = 30;
static constexpr int kT  = 3;
static constexpr int kH  = 512;
static constexpr int kHB = 512;
static constexpr int kE  = 6;
static constexpr int kEH = 512;
static constexpr int kTH = 512;
static constexpr int kMC = kB * kC;    // 15360 fused (b,c) rows
static constexpr int kGC = kT * kE;    // 18 real gate columns
static constexpr int kGL = 128;        // gate columns padded to one N-tile
static constexpr float kLossCoef = 0.01f;

// ---------------- GEMM tiling ----------------
static constexpr int TM = 64, TN = 128, TK = 32;  // 8 waves, each 16x64 of output

enum { MODE_F16 = 0, MODE_F32 = 1, MODE_EXPERT = 2, MODE_TOWER = 3 };

// =====================================================================
// B pre-pack: row-major f32 [K x N] -> fragment-tiled f16
//   dst[((kt*NB + nb)*32 + lane)*16 + pos],  NB = Npad/16
//   lane = (col%16) + 16*hl ; pos<8 -> K=kt*32+8*hl+pos ; pos>=8 -> +16
// =====================================================================
__global__ void pack_b(const float* __restrict__ src, _Float16* __restrict__ dst,
                       int K, int N, int Npad,
                       long long src_batch, long long dst_batch) {
  const int NB = Npad >> 4;
  const long long total = (long long)K * Npad;
  src += (long long)blockIdx.y * src_batch;
  dst += (long long)blockIdx.y * dst_batch;
  for (long long idx = (long long)blockIdx.x * blockDim.x + threadIdx.x;
       idx < total; idx += (long long)gridDim.x * blockDim.x) {
    const int pos  = (int)(idx & 15);
    const int lane = (int)((idx >> 4) & 31);
    const long long rest = idx >> 9;
    const int nb = (int)(rest % NB);
    const int kt = (int)(rest / NB);
    const int hl = lane >> 4;
    const int col = nb * 16 + (lane & 15);
    const int krel = (pos < 8) ? (8 * hl + pos) : (16 + 8 * hl + (pos - 8));
    const long long k = (long long)kt * 32 + krel;
    const float v = (col < N) ? src[k * N + col] : 0.f;
    dst[idx] = (_Float16)v;
  }
}

// w_gate [T][H][E] -> fragment-tiled [H x kGL] (cols t*E+e, zero-padded)
__global__ void pack_wgate(const float* __restrict__ wg, _Float16* __restrict__ dst) {
  const int idx = blockIdx.x * blockDim.x + threadIdx.x;
  if (idx >= kH * kGL) return;
  const int pos  = idx & 15;
  const int lane = (idx >> 4) & 31;
  const int rest = idx >> 9;
  const int nb = rest & 7;           // NB = 8
  const int kt = rest >> 3;
  const int hl = lane >> 4;
  const int col = nb * 16 + (lane & 15);
  const int krel = (pos < 8) ? (8 * hl + pos) : (16 + 8 * hl + (pos - 8));
  const int k = kt * 32 + krel;
  float v = 0.f;
  if (col < kGC) {
    const int t = col / kE, e = col % kE;
    v = wg[((size_t)t * kH + k) * kE + e];
  }
  dst[idx] = (_Float16)v;
}

// =====================================================================
// f16-WMMA GEMM. A double-buffered through LDS in fragment-major layout
// (one barrier per K-step); B read directly from the fragment-tiled
// global pack (L2-resident). Each wave: 1 A-frag -> 4 WMMA tiles (16x64).
// All N padded to multiples of TN -> no bounds checks.
// =====================================================================
template <bool A_IS_F32>
__global__ __launch_bounds__(256) void gemm_wmma(
    const void* __restrict__ Aptr, int lda,
    const _Float16* __restrict__ Bpack, int Npad,
    const float* __restrict__ bias,
    int K, int relu, int mode,
    void* __restrict__ outp, int ldc,
    // MODE_EXPERT
    const float* __restrict__ gatesp, int expert_e, int accumulate,
    float* __restrict__ Ybase, long long ystride,
    // MODE_TOWER
    const float* __restrict__ w2, float* __restrict__ logits, int t_idx)
{
  // double-buffered fragment-major A: [buf][16-row blk][lane][16 halves]
  __shared__ _Float16 As[2][4][32][16];

  const int tid  = threadIdx.x;
  const int lane = tid & 31;
  const int wave = tid >> 5;
  const int wm   = wave & 3;       // 4-way in M (64 rows)
  const int wn   = wave >> 2;      // 2-way in N (128 cols)
  const int hl   = lane >> 4;
  const int l15  = lane & 15;
  const int m0   = blockIdx.y * TM;
  const int n0   = blockIdx.x * TN;
  const int NB   = Npad >> 4;

  // A staging: thread covers 8 consecutive K of one row == one 16B LDS run
  const int ar    = tid >> 2;                   // 0..63 tile row
  const int ac    = (tid & 3) * 8;              // 0,8,16,24 K base
  const int ablk  = ar >> 4;
  const int alane = (ar & 15) + ((ac >> 3) & 1) * 16;
  const int apos  = (ac & 16) ? 8 : 0;

  auto load_tile = [&](int kk) -> v8h {
    v8h r;
    if constexpr (A_IS_F32) {
      const float* ap = (const float*)Aptr + (size_t)(m0 + ar) * lda + (kk + ac);
      float4 f0 = *(const float4*)(ap);
      float4 f1 = *(const float4*)(ap + 4);
      r[0] = (_Float16)f0.x; r[1] = (_Float16)f0.y;
      r[2] = (_Float16)f0.z; r[3] = (_Float16)f0.w;
      r[4] = (_Float16)f1.x; r[5] = (_Float16)f1.y;
      r[6] = (_Float16)f1.z; r[7] = (_Float16)f1.w;
    } else {
      r = *(const v8h*)((const _Float16*)Aptr + (size_t)(m0 + ar) * lda + (kk + ac));
    }
    return r;
  };

  const v8f zero = {0.f,0.f,0.f,0.f,0.f,0.f,0.f,0.f};
  v8f acc[4] = {zero, zero, zero, zero};
  const int cb0 = (n0 >> 4) + wn * 4;           // this wave's first col-block

  // prologue: stage tile 0
  {
    v8h pk = load_tile(0);
    *(v8h*)&As[0][ablk][alane][apos] = pk;
  }
  __syncthreads();

  const int niter = K / TK;
  for (int it = 0; it < niter; ++it) {
    const int cur = it & 1;
    const int k0 = it * TK;

    // stage next tile into the other buffer (overlaps with this tile's math)
    if (it + 1 < niter) {
      v8h nx = load_tile(k0 + TK);
      *(v8h*)&As[cur ^ 1][ablk][alane][apos] = nx;
      if (it + 2 < niter) {
        const char* pf = (const char*)Aptr +
            ((size_t)(m0 + ar) * lda + (k0 + 2 * TK + ac)) * (A_IS_F32 ? 4 : 2);
        __builtin_prefetch(pf, 0, 1);            // global_prefetch_b8
      }
    }

    // compute from current buffer
    const size_t ktbase = (size_t)(k0 >> 5) * NB;
    v16h afrag = *(const v16h*)&As[cur][wm][lane][0];     // 2x ds_load_b128
    #pragma unroll
    for (int hn = 0; hn < 4; ++hn) {
      v16h bf = *(const v16h*)(Bpack + ((ktbase + cb0 + hn) * 32 + lane) * 16);
      acc[hn] = __builtin_amdgcn_wmma_f32_16x16x32_f16(false, afrag, false, bf,
                                                       (short)0, acc[hn], false, false);
    }
    __syncthreads();   // single barrier per K-step
  }

  // ---- epilogue (C/D layout: vgpr i -> row i+8*hl, lane%16 -> col) ----
  #pragma unroll
  for (int i = 0; i < 8; ++i) {
    const int grow = m0 + wm * 16 + i + 8 * hl;
    float ga0 = 0.f, ga1 = 0.f, ga2 = 0.f;
    if (mode == MODE_EXPERT) {   // hoist per-row gate scalars out of the hn loop
      const float* gp = gatesp + (size_t)grow * kGC + expert_e;
      ga0 = gp[0]; ga1 = gp[kE]; ga2 = gp[2 * kE];
    }
    #pragma unroll
    for (int hn = 0; hn < 4; ++hn) {
      float val = acc[hn][i];
      const int gcol = n0 + wn * 64 + l15 + hn * 16;
      if (bias) val += bias[gcol];
      if (relu) val = fmaxf(val, 0.f);

      if (mode == MODE_TOWER) {
        float contrib = val * w2[gcol];
        contrib += __shfl_xor(contrib, 1, 32);
        contrib += __shfl_xor(contrib, 2, 32);
        contrib += __shfl_xor(contrib, 4, 32);
        contrib += __shfl_xor(contrib, 8, 32);
        if (l15 == 0) {
          const int bb = grow / kC, cc = grow % kC;
          atomicAdd(&logits[((size_t)bb * kT + t_idx) * kC + cc], contrib);
        }
      } else if (mode == MODE_F16) {
        ((_Float16*)outp)[(size_t)grow * ldc + gcol] = (_Float16)val;
      } else if (mode == MODE_F32) {
        ((float*)outp)[(size_t)grow * ldc + gcol] = val;
      } else {  // MODE_EXPERT: gated accumulate into per-task Y
        float* yp = Ybase + (size_t)grow * ldc + gcol;
        if (accumulate) {
          yp[0]           += ga0 * val;
          yp[ystride]     += ga1 * val;
          yp[2 * ystride] += ga2 * val;
        } else {
          yp[0]           = ga0 * val;
          yp[ystride]     = ga1 * val;
          yp[2 * ystride] = ga2 * val;
        }
      }
    }
  }
}

// =====================================================================
// Small helper kernels
// =====================================================================
__global__ void init_logits(const float* __restrict__ tb2, float* __restrict__ logits) {
  int idx = blockIdx.x * blockDim.x + threadIdx.x;
  if (idx >= kB * kT * kC) return;
  logits[idx] = tb2[(idx / kC) % kT];
}

// top-3-of-6 gating with softmax over kept experts; glog has row stride kGL
__global__ void gates_topk(const float* __restrict__ glog, float* __restrict__ gates) {
  int idx = blockIdx.x * blockDim.x + threadIdx.x;
  if (idx >= kMC * kT) return;
  const int row = idx / kT, t = idx % kT;
  const float* gl = glog + (size_t)row * kGL + t * kE;
  float v[kE], work[kE], g[kE];
  #pragma unroll
  for (int e = 0; e < kE; ++e) { v[e] = gl[e]; work[e] = v[e]; g[e] = 0.f; }
  float kv[3]; int ki[3];
  #pragma unroll
  for (int j = 0; j < 3; ++j) {            // stable argmax (ties -> lowest index)
    int mi = 0;
    #pragma unroll
    for (int e = 1; e < kE; ++e) if (work[e] > work[mi]) mi = e;
    kv[j] = v[mi]; ki[j] = mi; work[mi] = -3.4e38f;
  }
  const float mx = kv[0];
  float ev[3], s = 0.f;
  #pragma unroll
  for (int j = 0; j < 3; ++j) { ev[j] = expf(kv[j] - mx); s += ev[j]; }
  #pragma unroll
  for (int j = 0; j < 3; ++j) g[ki[j]] = ev[j] / s;
  float* go = gates + (size_t)row * kGC + t * kE;
  #pragma unroll
  for (int e = 0; e < kE; ++e) go[e] = g[e];
}

// single-block final reduction: labels-from-max, BCE, cv^2 aux, mean
__global__ __launch_bounds__(256) void loss_kernel(
    const float* __restrict__ scores, const float* __restrict__ logits,
    const float* __restrict__ gates, float* __restrict__ out) {
  const int tid = threadIdx.x;
  float accum = 0.f;
  for (int b = tid; b < kB; b += 256) {
    float tsum = 0.f, aux = 0.f;
    for (int t = 0; t < kT; ++t) {
      const float* sc = scores + ((size_t)b * kT + t) * kC;
      const float* lg = logits + ((size_t)b * kT + t) * kC;
      float mx = sc[0];
      for (int c = 1; c < kC; ++c) mx = fmaxf(mx, sc[c]);
      float bsum = 0.f;
      for (int c = 0; c < kC; ++c) {
        const float lab = (sc[c] == mx) ? 1.f : 0.f;
        const float l = lg[c];
        bsum += fmaxf(l, 0.f) - l * lab + log1pf(expf(-fabsf(l)));
      }
      tsum += bsum / (float)kC;
      float imp[kE] = {0.f, 0.f, 0.f, 0.f, 0.f, 0.f};
      for (int c = 0; c < kC; ++c) {
        const float* gp = gates + (size_t)(b * kC + c) * kGC + t * kE;
        #pragma unroll
        for (int e = 0; e < kE; ++e) imp[e] += gp[e];
      }
      float s = 0.f;
      #pragma unroll
      for (int e = 0; e < kE; ++e) s += imp[e];
      const float mean = s / (float)kE;
      float var = 0.f;
      #pragma unroll
      for (int e = 0; e < kE; ++e) { const float d = imp[e] - mean; var += d * d; }
      var /= (float)kE;
      aux += kLossCoef * var / (mean * mean + 1e-10f);
    }
    accum += tsum / (float)kT + aux;
  }
  __shared__ float red[256];
  red[tid] = accum; __syncthreads();
  for (int s = 128; s > 0; s >>= 1) {
    if (tid < s) red[tid] += red[tid + s];
    __syncthreads();
  }
  if (tid == 0) out[0] = red[0] / (float)kB;
}

// =====================================================================
// Host orchestration
// =====================================================================
extern "C" void kernel_launch(void* const* d_in, const int* in_sizes, int n_in,
                              void* d_out, int out_size, void* d_ws, size_t ws_size,
                              hipStream_t stream) {
  const float* x      = (const float*)d_in[0];   // [B,C,H]
  const float* scores = (const float*)d_in[1];   // [B,T,C]
  const float* fc1_w  = (const float*)d_in[2];
  const float* fc1_b  = (const float*)d_in[3];
  const float* fc2_w  = (const float*)d_in[4];
  const float* fc2_b  = (const float*)d_in[5];
  const float* w_gate = (const float*)d_in[6];   // [T,H,E]
  const float* ew1    = (const float*)d_in[7];   // [E,H,EH]
  const float* eb1    = (const float*)d_in[8];
  const float* ew2    = (const float*)d_in[9];   // [E,EH,H]
  const float* eb2    = (const float*)d_in[10];
  const float* tw1    = (const float*)d_in[11];  // [T,H,TH]
  const float* tb1    = (const float*)d_in[12];
  const float* tw2    = (const float*)d_in[13];  // [T,TH]
  const float* tb2    = (const float*)d_in[14];
  float* out = (float*)d_out;
  (void)in_sizes; (void)n_in; (void)out_size; (void)ws_size; // k hardcoded to 3

  // ---- carve workspace (256B aligned) ----
  char* w = (char*)d_ws;
  auto carve = [&](size_t bytes) -> void* {
    void* p = (void*)w;
    w += (bytes + 255) & ~(size_t)255;
    return p;
  };
  _Float16* buf0   = (_Float16*)carve((size_t)kMC * kH * 2);      // h (f16)
  _Float16* buf1   = (_Float16*)carve((size_t)kMC * kH * 2);      // fc1-out / eh (f16)
  _Float16* fc1w16 = (_Float16*)carve((size_t)kH * kHB * 2);      // fragment-tiled
  _Float16* fc2w16 = (_Float16*)carve((size_t)kHB * kH * 2);
  _Float16* wg16   = (_Float16*)carve((size_t)kH * kGL * 2);
  _Float16* ew1_16 = (_Float16*)carve((size_t)kE * kH * kEH * 2);
  _Float16* ew2_16 = (_Float16*)carve((size_t)kE * kEH * kH * 2);
  _Float16* tw1_16 = (_Float16*)carve((size_t)kT * kH * kTH * 2);
  float*    glog   = (float*)carve((size_t)kMC * kGL * 4);
  float*    gates  = (float*)carve((size_t)kMC * kGC * 4);
  float*    Y      = (float*)carve((size_t)kT * kMC * kH * 4);    // gated expert mix
  float*    logits = (float*)carve((size_t)kB * kT * kC * 4);

  const dim3 blk(256);
  auto grid_of = [](int M, int Npad) { return dim3(Npad / TN, M / TM); };
  const long long ystr = (long long)kMC * kH;
  const long long wsz  = (long long)kH * kH;

  // ---- pack all weights into fragment-tiled f16 (one-time) ----
  pack_b<<<dim3(256, 1), blk, 0, stream>>>(fc1_w, fc1w16, kH, kHB, kHB, 0, 0);
  pack_b<<<dim3(256, 1), blk, 0, stream>>>(fc2_w, fc2w16, kHB, kH, kH, 0, 0);
  pack_wgate<<<(kH * kGL + 255) / 256, blk, 0, stream>>>(w_gate, wg16);
  pack_b<<<dim3(256, kE), blk, 0, stream>>>(ew1, ew1_16, kH, kEH, kEH, wsz, wsz);
  pack_b<<<dim3(256, kE), blk, 0, stream>>>(ew2, ew2_16, kEH, kH, kH, wsz, wsz);
  pack_b<<<dim3(256, kT), blk, 0, stream>>>(tw1, tw1_16, kH, kTH, kTH, wsz, wsz);

  // ---- shared bottom: buf1 = relu(x@fc1+b); buf0 = buf1@fc2+b ----
  gemm_wmma<true><<<grid_of(kMC, kHB), blk, 0, stream>>>(
      x, kH, fc1w16, kHB, fc1_b, kH, 1, MODE_F16, buf1, kHB,
      nullptr, 0, 0, nullptr, 0, nullptr, nullptr, 0);
  gemm_wmma<false><<<grid_of(kMC, kH), blk, 0, stream>>>(
      buf1, kHB, fc2w16, kH, fc2_b, kHB, 0, MODE_F16, buf0, kH,
      nullptr, 0, 0, nullptr, 0, nullptr, nullptr, 0);

  // ---- gate logits [15360, 128-padded] then top-3 softmax gates ----
  gemm_wmma<false><<<grid_of(kMC, kGL), blk, 0, stream>>>(
      buf0, kH, wg16, kGL, nullptr, kH, 0, MODE_F32, glog, kGL,
      nullptr, 0, 0, nullptr, 0, nullptr, nullptr, 0);
  gates_topk<<<(kMC * kT + 255) / 256, blk, 0, stream>>>(glog, gates);

  // ---- experts: buf1 = relu(h@W1[e]+b); Y[t] (+)= g[b,t,c,e]*(buf1@W2[e]+b) ----
  for (int e = 0; e < kE; ++e) {
    gemm_wmma<false><<<grid_of(kMC, kEH), blk, 0, stream>>>(
        buf0, kH, ew1_16 + (size_t)e * wsz, kEH, eb1 + (size_t)e * kEH,
        kH, 1, MODE_F16, buf1, kEH,
        nullptr, 0, 0, nullptr, 0, nullptr, nullptr, 0);
    gemm_wmma<false><<<grid_of(kMC, kH), blk, 0, stream>>>(
        buf1, kEH, ew2_16 + (size_t)e * wsz, kH, eb2 + (size_t)e * kH,
        kEH, 0, MODE_EXPERT, nullptr, kH,
        gates, e, (e > 0) ? 1 : 0, Y, ystr, nullptr, nullptr, 0);
  }

  // ---- towers: logits = tb2 + sum_f relu(Y[t]@W1[t]+b1)*w2 (fused reduce) ----
  init_logits<<<(kB * kT * kC + 255) / 256, blk, 0, stream>>>(tb2, logits);
  for (int t = 0; t < kT; ++t) {
    gemm_wmma<true><<<grid_of(kMC, kTH), blk, 0, stream>>>(
        Y + (size_t)t * ystr, kH, tw1_16 + (size_t)t * wsz, kTH,
        tb1 + (size_t)t * kTH, kH, 1, MODE_TOWER, nullptr, kTH,
        nullptr, 0, 0, nullptr, 0, tw2 + (size_t)t * kTH, logits, t);
  }

  // ---- scalar loss ----
  loss_kernel<<<1, blk, 0, stream>>>(scores, logits, gates, out);
}
